// GRUModel_25391846654518
// MI455X (gfx1250) — matrix operand order repaired and checked by reading, not compiled, not run on hardware
//
#include <hip/hip_runtime.h>
#include <math.h>

// ---------------------------------------------------------------------------
// GRU model on MI455X (gfx1250, wave32, WMMA bf16 16x16x32, f32 accumulate)
//   B=256, T=512, D=300, U=64, C=45, 3U=192
// Pipeline:
//   k0: convert/pre-swizzle weights -> bf16 WMMA B-fragment blocks in ws
//   k1: mx = x @ kernel + bias[0]          (M=131072, K=300->320, N=192)
//   k2: fused GRU scan, 16 WGs (one per 16-row batch tile), h in LDS
//   k3: logits = hs @ dense_w + b, fused row softmax (N pad 45->48)
// ---------------------------------------------------------------------------

typedef __attribute__((ext_vector_type(16))) __bf16 v16bf;
typedef __attribute__((ext_vector_type(8)))  __bf16 v8bf;
typedef __attribute__((ext_vector_type(8)))  float  v8f;

#define BSZ 256
#define TSZ 512
#define DSZ 300
#define USZ 64
#define CSZ 45
#define N3U 192

// B-fragment block: one 16x16x32 WMMA B tile = 512 bf16, laid out so that
// lane L reads its 16 bf16 (32B) contiguously at block*512 + L*16.
// Mapping: lane = n_local + 16*((kloc>>3)&1), elem = ((kloc>>4)<<3) + (kloc&7), kloc in [0,32).

// workspace layout (bytes)
#define OFF_WK   0u                    // 10*12*512 bf16 = 122880 B
#define OFF_WREC 122880u               //  2*12*512 bf16 =  24576 B
#define OFF_WD   147456u               //  2* 3*512 bf16 =   6144 B
#define OFF_MX   153856u               // 131072*192 f32 = 100663296 B
#define OFF_HS   100817152u            // 131072*64 bf16 =  16777216 B

// ---------------------------------------------------------------------------
// k0: weight conversion + B-fragment swizzle (bf16, zero-padded)
// ---------------------------------------------------------------------------
__global__ __launch_bounds__(256) void gru_convert_weights(
    const float* __restrict__ Wk, const float* __restrict__ Wrec,
    const float* __restrict__ Wd,
    __bf16* __restrict__ wk_f, __bf16* __restrict__ wrec_f,
    __bf16* __restrict__ wd_f) {
  const int WK_N = 10 * 12 * 512;   // 61440
  const int WR_N = 2 * 12 * 512;    // 12288
  const int WD_N = 2 * 3 * 512;     //  3072
  int idx = blockIdx.x * 256 + threadIdx.x;
  const float* src; __bf16* dst; int K, N, ldn, NT, li;
  if (idx < WK_N)               { src = Wk;   dst = wk_f;   K = DSZ; N = N3U; ldn = N3U; NT = 12; li = idx; }
  else if (idx < WK_N + WR_N)   { src = Wrec; dst = wrec_f; K = USZ; N = N3U; ldn = N3U; NT = 12; li = idx - WK_N; }
  else if (idx < WK_N + WR_N + WD_N) { src = Wd; dst = wd_f; K = USZ; N = CSZ; ldn = CSZ; NT = 3; li = idx - WK_N - WR_N; }
  else return;
  int blk  = li >> 9, r = li & 511;
  int lane = r >> 4, elem = r & 15;
  int kt = blk / NT, nt = blk % NT;
  int kloc = ((elem >> 3) << 4) + ((lane >> 4) << 3) + (elem & 7);        // 0..31
  int k = kt * 32 + kloc;
  int n = nt * 16 + (lane & 15);
  float v = (k < K && n < N) ? src[k * ldn + n] : 0.0f;
  dst[li] = (__bf16)v;
}

// ---------------------------------------------------------------------------
// k1: mx = x @ kernel + bias0.  1024 WGs x 256 thr (8 waves); wave -> 16-row
// M tile, full N=192 (12 accum tiles), K loop 10 x 32 (D padded to 320).
// ---------------------------------------------------------------------------
__global__ __launch_bounds__(256) void gru_proj_gemm(
    const float* __restrict__ x, const __bf16* __restrict__ wk_f,
    const float* __restrict__ bias0, float* __restrict__ mx) {
  const int lane = threadIdx.x & 31;
  const int wave = threadIdx.x >> 5;
  const int mt   = blockIdx.x * 8 + wave;        // 0..8191
  const int hi   = lane >> 4;                    // lane half
  const int nl   = lane & 15;
  const float* xrow = x + (long)(mt * 16 + nl) * DSZ;

  v8f acc[12] = {};
  for (int kt = 0; kt < 10; ++kt) {
    // A fragment: 16x32 bf16. lane<16: K {0..7,16..23}; lane>=16: {8..15,24..31}
    v16bf a;
#pragma unroll
    for (int g = 0; g < 2; ++g) {
      int ks = kt * 32 + g * 16 + hi * 8;        // 8 consecutive K
      if (ks + 8 <= DSZ) {
        const float4* p = (const float4*)(xrow + ks);
        float4 f0 = p[0], f1 = p[1];
        a[g * 8 + 0] = (__bf16)f0.x; a[g * 8 + 1] = (__bf16)f0.y;
        a[g * 8 + 2] = (__bf16)f0.z; a[g * 8 + 3] = (__bf16)f0.w;
        a[g * 8 + 4] = (__bf16)f1.x; a[g * 8 + 5] = (__bf16)f1.y;
        a[g * 8 + 6] = (__bf16)f1.z; a[g * 8 + 7] = (__bf16)f1.w;
      } else {
#pragma unroll
        for (int e = 0; e < 8; ++e) {
          int k = ks + e;
          a[g * 8 + e] = (__bf16)((k < DSZ) ? xrow[k] : 0.0f);
        }
      }
    }
    const __bf16* bbase = wk_f + (long)(kt * 12) * 512 + lane * 16;
#pragma unroll
    for (int nt = 0; nt < 12; ++nt) {
      v16bf b = *(const v16bf*)(bbase + nt * 512);
      acc[nt] = __builtin_amdgcn_wmma_f32_16x16x32_bf16(
          false, a, false, b, (short)0, acc[nt], false, false);
    }
  }
  // C/D layout: VGPR j -> row (j + 8*hi), col = nl within tile
  const int rbase = mt * 16 + hi * 8;
#pragma unroll
  for (int nt = 0; nt < 12; ++nt) {
    int n = nt * 16 + nl;
    float bv = bias0[n];
#pragma unroll
    for (int j = 0; j < 8; ++j)
      mx[(long)(rbase + j) * N3U + n] = acc[nt][j] + bv;
  }
}

// ---------------------------------------------------------------------------
// k2: fused sequential GRU scan.  16 WGs (one per 16-row batch tile),
// 128 thr = 4 waves.  Waves 0..2 each compute one gate's 16x64 mi slab via
// 8 WMMAs/step from the LDS-resident h A-fragment; all threads do the f32
// gate math and rebuild the h fragment in LDS.  hs written as bf16.
// ---------------------------------------------------------------------------
__global__ __launch_bounds__(128) void gru_scan(
    const float* __restrict__ mx, const __bf16* __restrict__ wrec_f,
    const float* __restrict__ rec_bias, __bf16* __restrict__ hs) {
  __shared__ __bf16 h_frag[2 * 512];     // h as WMMA A fragments (2 k-tiles)
  __shared__ float  mi_s[16 * N3U];      // h @ Wrec + rec_bias
  __shared__ float  mx_s[16 * N3U];      // staged mx tile for this t
  __shared__ float  h_s[16 * USZ];       // h in f32

  const int tid   = threadIdx.x;
  const int lane  = tid & 31;
  const int wave  = tid >> 5;
  const int btile = blockIdx.x;          // 0..15

  for (int i = tid; i < 1024; i += 128) h_frag[i] = (__bf16)0.0f;
  for (int i = tid; i < 16 * USZ; i += 128) h_s[i] = 0.0f;

  // Persistent recurrent-weight B fragments (4 N tiles x 2 K tiles per wave)
  v16bf wb[8];
  if (wave < 3) {
    const __bf16* base = wrec_f + lane * 16;
#pragma unroll
    for (int i = 0; i < 4; ++i)
#pragma unroll
      for (int kt = 0; kt < 2; ++kt)
        wb[i * 2 + kt] = *(const v16bf*)(base + (long)(kt * 12 + wave * 4 + i) * 512);
  }
  __syncthreads();

  const int r_ld  = tid >> 3;            // mx stage: row 0..15
  const int c_ld  = (tid & 7) * 24;      // 24 floats per thread
  const int r_ew  = tid >> 3;            // elementwise: row
  const int u0    = (tid & 7) * 8;       // 8 consecutive u per thread

  for (int t = 0; t < TSZ; ++t) {
    // stage mx[btile, :, t, :] -> LDS; prefetch next timestep
    {
      const float* src = mx + ((long)(btile * 16 + r_ld) * TSZ + t) * N3U + c_ld;
      const float4* s4 = (const float4*)src;
      float4* d4 = (float4*)&mx_s[r_ld * N3U + c_ld];
#pragma unroll
      for (int q = 0; q < 6; ++q) d4[q] = s4[q];
      if (t + 1 < TSZ) __builtin_prefetch(src + N3U, 0, 3);
    }
    __syncthreads();

    // mi = h @ Wrec + rec_bias  (waves 0..2, one gate each)
    if (wave < 3) {
      v8f acc[4] = {};
#pragma unroll
      for (int kt = 0; kt < 2; ++kt) {
        v16bf a = *(const v16bf*)&h_frag[kt * 512 + lane * 16];
#pragma unroll
        for (int i = 0; i < 4; ++i)
          acc[i] = __builtin_amdgcn_wmma_f32_16x16x32_bf16(
              false, a, false, wb[i * 2 + kt], (short)0, acc[i], false, false);
      }
      const int nl = lane & 15, hi = lane >> 4;
#pragma unroll
      for (int i = 0; i < 4; ++i) {
        int n = wave * 64 + i * 16 + nl;
        float rb = rec_bias[n];
#pragma unroll
        for (int j = 0; j < 8; ++j)
          mi_s[(j + hi * 8) * N3U + n] = acc[i][j] + rb;
      }
    }
    __syncthreads();

    // gate math + h update (all 128 threads, 8 u-elements each)
    {
      v8bf hb;
      float hn[8];
#pragma unroll
      for (int i = 0; i < 8; ++i) {
        int u = u0 + i;
        float z  = mx_s[r_ew * N3U + u]        + mi_s[r_ew * N3U + u];
        z = 1.0f / (1.0f + __expf(-z));
        float rr = mx_s[r_ew * N3U + 64 + u]   + mi_s[r_ew * N3U + 64 + u];
        rr = 1.0f / (1.0f + __expf(-rr));
        float hh = mx_s[r_ew * N3U + 128 + u] + rr * mi_s[r_ew * N3U + 128 + u];
        hh = fmaxf(hh, 0.0f);
        float h  = h_s[r_ew * USZ + u];
        float v  = z * h + (1.0f - z) * hh;
        hn[i] = v;
        hb[i] = (__bf16)v;
      }
#pragma unroll
      for (int i = 0; i < 8; ++i) h_s[r_ew * USZ + u0 + i] = hn[i];
      // rebuild h A-fragment: u0 multiple of 8 -> one contiguous 16B slot
      int kt = u0 >> 5;
      int ln = r_ew + (((u0 >> 3) & 1) << 4);
      int e0 = ((u0 >> 4) & 1) * 8;
      *(v8bf*)&h_frag[kt * 512 + ln * 16 + e0] = hb;
      *(v8bf*)(hs + ((long)(btile * 16 + r_ew) * TSZ + t) * USZ + u0) = hb;
    }
    __syncthreads();
  }
}

// ---------------------------------------------------------------------------
// k3: logits = hs @ dense_w + b, fused softmax over C=45 (padded to 48;
// pad logits = -inf).  Row reductions via 16-lane shfl_xor halves (wave32:
// C/D layout holds one row's N in one 16-lane half).
// ---------------------------------------------------------------------------
__global__ __launch_bounds__(256) void gru_dense_softmax(
    const __bf16* __restrict__ hs, const __bf16* __restrict__ wd_f,
    const float* __restrict__ dense_b, float* __restrict__ out) {
  const int lane = threadIdx.x & 31;
  const int wave = threadIdx.x >> 5;
  const int mt   = blockIdx.x * 8 + wave;        // 0..8191
  const int hi   = lane >> 4;
  const int nl   = lane & 15;
  const __bf16* arow = hs + (long)(mt * 16 + nl) * USZ;

  v8f acc[3] = {};
#pragma unroll
  for (int kt = 0; kt < 2; ++kt) {
    v16bf a;
#pragma unroll
    for (int g = 0; g < 2; ++g) {
      int ks = kt * 32 + g * 16 + hi * 8;
      v8bf seg = *(const v8bf*)(arow + ks);
#pragma unroll
      for (int e = 0; e < 8; ++e) a[g * 8 + e] = seg[e];
    }
    const __bf16* bbase = wd_f + (long)(kt * 3) * 512 + lane * 16;
#pragma unroll
    for (int nt = 0; nt < 3; ++nt) {
      v16bf b = *(const v16bf*)(bbase + nt * 512);
      acc[nt] = __builtin_amdgcn_wmma_f32_16x16x32_bf16(
          false, a, false, b, (short)0, acc[nt], false, false);
    }
  }

  float l[3][8];
#pragma unroll
  for (int nt = 0; nt < 3; ++nt) {
    int n = nt * 16 + nl;
    float bv = (n < CSZ) ? dense_b[n] : -INFINITY;
#pragma unroll
    for (int j = 0; j < 8; ++j)
      l[nt][j] = (n < CSZ) ? (acc[nt][j] + bv) : -INFINITY;
  }
#pragma unroll
  for (int j = 0; j < 8; ++j) {
    float m = fmaxf(fmaxf(l[0][j], l[1][j]), l[2][j]);
#pragma unroll
    for (int off = 8; off > 0; off >>= 1)
      m = fmaxf(m, __shfl_xor(m, off, 32));      // stays inside 16-lane half
    float e0 = __expf(l[0][j] - m);
    float e1 = __expf(l[1][j] - m);
    float e2 = __expf(l[2][j] - m);
    float s = e0 + e1 + e2;
#pragma unroll
    for (int off = 8; off > 0; off >>= 1)
      s += __shfl_xor(s, off, 32);
    float inv = 1.0f / s;
    int row = mt * 16 + j + hi * 8;
    float ev[3] = {e0, e1, e2};
#pragma unroll
    for (int nt = 0; nt < 3; ++nt) {
      int n = nt * 16 + nl;
      if (n < CSZ) out[(long)row * CSZ + n] = ev[nt] * inv;
    }
  }
}

// ---------------------------------------------------------------------------
extern "C" void kernel_launch(void* const* d_in, const int* in_sizes, int n_in,
                              void* d_out, int out_size, void* d_ws, size_t ws_size,
                              hipStream_t stream) {
  const float* x      = (const float*)d_in[0];   // [B,T,D]
  const float* Wk     = (const float*)d_in[1];   // [D,3U]
  const float* Wrec   = (const float*)d_in[2];   // [U,3U]
  const float* bias   = (const float*)d_in[3];   // [2,3U]
  const float* Wd     = (const float*)d_in[4];   // [U,C]
  const float* db     = (const float*)d_in[5];   // [C]

  char* ws = (char*)d_ws;
  __bf16* wk_f   = (__bf16*)(ws + OFF_WK);
  __bf16* wrec_f = (__bf16*)(ws + OFF_WREC);
  __bf16* wd_f   = (__bf16*)(ws + OFF_WD);
  float*  mx     = (float*)(ws + OFF_MX);
  __bf16* hs     = (__bf16*)(ws + OFF_HS);

  gru_convert_weights<<<300, 256, 0, stream>>>(Wk, Wrec, Wd, wk_f, wrec_f, wd_f);
  gru_proj_gemm<<<1024, 256, 0, stream>>>(x, wk_f, bias, mx);
  gru_scan<<<16, 128, 0, stream>>>(mx, wrec_f, bias + N3U, hs);
  gru_dense_softmax<<<1024, 256, 0, stream>>>(hs, wd_f, db, (float*)d_out);
}
